// Featurizer_16990890623129
// MI455X (gfx1250) — matrix operand order repaired and checked by reading, not compiled, not run on hardware
//
#include <hip/hip_runtime.h>

#define N_RES   30000
#define N_ATOMS 6
#define N_EDGES 300000
#define HIDDEN  128
#define FEPS    1e-6f

typedef unsigned short u16;
typedef __attribute__((ext_vector_type(16))) __bf16 v16bf;
typedef __attribute__((ext_vector_type(8)))  float  v8f;

union Frag { u16 u[16]; v16bf v; };

__device__ __forceinline__ u16 f2bf(float f) {
    unsigned u = __float_as_uint(f);
    unsigned r = u + 0x7FFFu + ((u >> 16) & 1u);   // round-to-nearest-even
    return (u16)(r >> 16);
}
__device__ __forceinline__ float signf(float x) {
    return (x > 0.f) ? 1.f : ((x < 0.f) ? -1.f : 0.f);
}

struct F3 { float x, y, z; };
__device__ __forceinline__ F3 sub3(F3 a, F3 b) { return {a.x-b.x, a.y-b.y, a.z-b.z}; }
__device__ __forceinline__ float dot3(F3 a, F3 b) { return a.x*b.x + a.y*b.y + a.z*b.z; }
__device__ __forceinline__ F3 cross3(F3 a, F3 b) {
    return {a.y*b.z - a.z*b.y, a.z*b.x - a.x*b.z, a.x*b.y - a.y*b.x};
}
__device__ __forceinline__ F3 norm3(F3 a) {           // matches _normalize: x / max(|x|,1e-12)
    float n = fmaxf(sqrtf(dot3(a, a)), 1e-12f);
    return {a.x/n, a.y/n, a.z/n};
}
__device__ __forceinline__ F3 ld3(const float* p) { return {p[0], p[1], p[2]}; }

// Per-residue local frame Q = [bvec | nvec | bvec x nvec], zeroed when q_valid is false.
__device__ __forceinline__ bool computeQ(const float* __restrict__ cc,
                                         const int* __restrict__ batch, int r,
                                         F3& qb, F3& qn, F3& qc) {
    bool bnd  = (r > 0)         && (batch[r]   != batch[r-1]);
    bool bndn = (r < N_RES - 1) && (batch[r+1] != batch[r]);
    bool valid = (r > 0) && (r < N_RES - 1) && !bnd && !bndn;
    if (!valid) { qb = {0,0,0}; qn = {0,0,0}; qc = {0,0,0}; return false; }
    F3 p0 = ld3(cc + 3*(r-1));
    F3 p1 = ld3(cc + 3*r);
    F3 p2 = ld3(cc + 3*(r+1));
    F3 du0 = norm3(sub3(p1, p0));
    F3 du1 = norm3(sub3(p2, p1));
    qb = norm3(sub3(du0, du1));
    qn = norm3(cross3(du0, du1));
    qc = cross3(qb, qn);
    return true;
}

// ---------------------------------------------------------------------------
// Weight prep: f32 -> bf16, zero-pad K (44->64, 688->704), and swizzle into
// WMMA B-fragment order: [ntile][kstep][lane][elem16] where lane<16 covers
// K = kstep*32 + 0..15 and lane>=16 covers K = kstep*32 + 16..31 (contiguous k order, matched by the A loads).
// ---------------------------------------------------------------------------
__global__ void __launch_bounds__(256)
convert_weights(const float* __restrict__ nodeW, const float* __restrict__ edgeW,
                u16* __restrict__ nodeWsw, u16* __restrict__ edgeWsw) {
    int idx = blockIdx.x * 256 + threadIdx.x;
    const int NODE_TOT = 8 * 2 * 32 * 16;    // 8192
    const int EDGE_TOT = 8 * 22 * 32 * 16;   // 90112
    if (idx < NODE_TOT) {
        int i = idx & 15, lane = (idx >> 4) & 31, rest = idx >> 9;
        int ks = rest & 1, nt = rest >> 1;
        int k   = ks * 32 + (lane >> 4) * 16 + i;
        int col = nt * 16 + (lane & 15);
        nodeWsw[idx] = (k < 44) ? f2bf(nodeW[k * HIDDEN + col]) : (u16)0;
    } else if (idx < NODE_TOT + EDGE_TOT) {
        int j = idx - NODE_TOT;
        int i = j & 15, lane = (j >> 4) & 31, rest = j >> 9;
        int ks = rest % 22, nt = rest / 22;
        int k   = ks * 32 + (lane >> 4) * 16 + i;
        int col = nt * 16 + (lane & 15);
        edgeWsw[j] = (k < 688) ? f2bf(edgeW[k * HIDDEN + col]) : (u16)0;
    }
}

// ---------------------------------------------------------------------------
// Node pipeline: 16 residues per workgroup (4 wave32).
// features(44, pad 64) -> LDS bf16 -> WMMA 16x64 @ 64x128 -> bias+LN -> out
// ---------------------------------------------------------------------------
__global__ void __launch_bounds__(128)
node_kernel(const float* __restrict__ coords, const float* __restrict__ cc,
            const int* __restrict__ batch, const u16* __restrict__ Wsw,
            const float* __restrict__ bias, const float* __restrict__ lnw,
            const float* __restrict__ lnb, float* __restrict__ out) {
    __shared__ u16   sfeat[16][64];
    __shared__ float sout[16][HIDDEN];
    __shared__ float smu[16], srstd[16];
    const int tid = threadIdx.x;
    const int r0  = blockIdx.x * 16;
    const long L  = (long)N_RES * N_ATOMS;

    if (tid < 16) {
        int r = r0 + tid;
        float feat[44];
        // Xf[6r-1 .. 6r+7]
        F3 P[9];
        #pragma unroll
        for (int p = 0; p < 9; p++) {
            long f = (long)6 * r - 1 + p;
            P[p] = (f >= 0 && f < L) ? ld3(coords + 3 * f) : F3{0.f, 0.f, 0.f};
        }
        F3 Ul[8];
        #pragma unroll
        for (int p = 0; p < 8; p++) Ul[p] = norm3(sub3(P[p+1], P[p]));
        bool bnd  = (r > 0)         && (batch[r]   != batch[r-1]);
        bool bndn = (r < N_RES - 1) && (batch[r+1] != batch[r]);
        #pragma unroll
        for (int a = 0; a < 6; a++) {
            long j = (long)6 * r + a;
            // dihedral: sin(arccos c * s) = s*sqrt(1-c^2), cos unchanged
            bool dm = (j > 0) && (j < L - 2) && !(bnd && a == 0) && !(bndn && a >= 4);
            F3 c1 = norm3(cross3(Ul[a],     Ul[a + 1]));
            F3 c2 = norm3(cross3(Ul[a + 1], Ul[a + 2]));
            float cd = fminf(fmaxf(dot3(c1, c2), -1.f + FEPS), 1.f - FEPS);
            float sd = signf(dot3(c2, Ul[a])) * sqrtf(fmaxf(1.f - cd * cd, 0.f));
            feat[2*a]     = dm ? sd : 0.f;
            feat[2*a + 1] = dm ? cd : 0.f;
            // bond angle
            bool am = (j > 0) && (j < L - 1) && !(bnd && a == 0) && !(bndn && a == 5);
            F3 d0 = norm3(sub3(P[a],     P[a + 1]));
            F3 d1 = norm3(sub3(P[a + 2], P[a + 1]));
            float ca = dot3(d0, d1);
            float sa = sqrtf(fmaxf(1.f - ca * ca + FEPS, 0.f));
            feat[12 + 2*a] = am ? sa : 0.f;
            feat[13 + 2*a] = am ? ca : 0.f;
        }
        F3 qb, qn, qc;
        computeQ(cc, batch, r, qb, qn, qc);
        F3 cx = ld3(cc + 3 * r);
        int idx = 0;
        #pragma unroll
        for (int a = 0; a < 6; a++) {
            if (a == 1) continue;                 // CENTRAL_IDX
            F3 d = sub3(P[a + 1], cx);            // P[a+1] == Xf[6r+a]
            float dd = dot3(d, d);
            feat[24 + idx] = logf(sqrtf(dd + FEPS));
            F3 u = norm3(d);
            feat[29 + idx*3 + 0] = dot3(u, qb);
            feat[29 + idx*3 + 1] = dot3(u, qn);
            feat[29 + idx*3 + 2] = dot3(u, qc);
            idx++;
        }
        for (int k = 0; k < 44; k++) sfeat[tid][k] = f2bf(feat[k]);
        for (int k = 44; k < 64; k++) sfeat[tid][k] = 0;
    }
    __syncthreads();

    {   // WMMA GEMM: 16x64 (bf16) @ 64x128 (bf16) -> f32
        const int lane = tid & 31, wave = tid >> 5;
        const int m = lane & 15, g = lane >> 4;
        for (int t = 0; t < 2; t++) {
            const int n0 = wave * 32 + t * 16;
            const int ntile = n0 >> 4;
            v8f c = {};
            for (int ks = 0; ks < 2; ks++) {
                Frag a, b;
                const u16* ap = &sfeat[m][ks * 32];
                #pragma unroll
                for (int i = 0; i < 8; i++) { a.u[i] = ap[g*16 + i]; a.u[8+i] = ap[g*16 + 8 + i]; }
                const u16* bp = &Wsw[((ntile * 2 + ks) * 32 + lane) * 16];
                #pragma unroll
                for (int i = 0; i < 16; i++) b.u[i] = bp[i];
                c = __builtin_amdgcn_wmma_f32_16x16x32_bf16(false, a.v, false, b.v,
                                                            (short)0, c, false, false);
            }
            #pragma unroll
            for (int v = 0; v < 8; v++) {
                int row = g ? (v + 8) : v;
                int col = n0 + m;
                sout[row][col] = c[v] + bias[col];
            }
        }
    }
    __syncthreads();

    if (tid < 16) {
        float s = 0.f, ss = 0.f;
        for (int j = 0; j < HIDDEN; j++) { float y = sout[tid][j]; s += y; ss += y * y; }
        float mu  = s * (1.f / HIDDEN);
        float var = ss * (1.f / HIDDEN) - mu * mu;
        smu[tid] = mu; srstd[tid] = 1.f / sqrtf(fmaxf(var, 0.f) + 1e-5f);
    }
    __syncthreads();
    {
        int row = tid >> 3, part = tid & 7;
        float mu = smu[row], rs = srstd[row];
        long base = (long)(r0 + row) * HIDDEN;
        #pragma unroll
        for (int jj = 0; jj < 16; jj++) {
            int j = part * 16 + jj;
            out[base + j] = (sout[row][j] - mu) * rs * lnw[j] + lnb[j];
        }
    }
}

// ---------------------------------------------------------------------------
// Edge pipeline: 16 edges per workgroup (4 wave32).
// features(688, pad 704) -> LDS bf16 -> WMMA 16x704 @ 704x128 -> bias+LN -> out
// ---------------------------------------------------------------------------
__global__ void __launch_bounds__(128)
edge_kernel(const float* __restrict__ coords, const float* __restrict__ cc,
            const int* __restrict__ batch, const int* __restrict__ eidx,
            const u16* __restrict__ Wsw, const float* __restrict__ bias,
            const float* __restrict__ lnw, const float* __restrict__ lnb,
            float* __restrict__ out) {
    __shared__ u16   sfeat[16][704];
    __shared__ float sQt[16][9];
    __shared__ float sout[16][HIDDEN];
    __shared__ float smu[16], srstd[16];
    const int tid = threadIdx.x;
    const int e0  = blockIdx.x * 16;

    if (tid < 16) {                                // quaternion + target frame
        int e   = e0 + tid;
        int src = eidx[e];
        int tgt = eidx[N_EDGES + e];
        F3 tb, tn, tc, sb, sn, sc;
        bool vt = computeQ(cc, batch, tgt, tb, tn, tc);
        bool vs = computeQ(cc, batch, src, sb, sn, sc);
        F3 ti[3] = {tb, tn, tc};
        F3 si[3] = {sb, sn, sc};
        float R[3][3];
        #pragma unroll
        for (int i = 0; i < 3; i++)
            #pragma unroll
            for (int j = 0; j < 3; j++) R[i][j] = dot3(ti[i], si[j]);
        float Rxx = R[0][0], Ryy = R[1][1], Rzz = R[2][2];
        float m0 = 0.5f * sqrtf(fabsf(1.f + Rxx - Ryy - Rzz));
        float m1 = 0.5f * sqrtf(fabsf(1.f - Rxx + Ryy - Rzz));
        float m2 = 0.5f * sqrtf(fabsf(1.f - Rxx - Ryy + Rzz));
        float x  = signf(R[2][1] - R[1][2]) * m0;
        float y  = signf(R[0][2] - R[2][0]) * m1;
        float z  = signf(R[1][0] - R[0][1]) * m2;
        float w  = 0.5f * sqrtf(fmaxf(1.f + Rxx + Ryy + Rzz, 0.f));
        float nq = fmaxf(sqrtf(x*x + y*y + z*z + w*w), 1e-12f);
        bool both = vs && vt;
        sfeat[tid][0] = f2bf(both ? x / nq : 0.f);
        sfeat[tid][1] = f2bf(both ? y / nq : 0.f);
        sfeat[tid][2] = f2bf(both ? z / nq : 0.f);
        sfeat[tid][3] = f2bf(both ? w / nq : 0.f);
        sQt[tid][0] = tb.x; sQt[tid][1] = tb.y; sQt[tid][2] = tb.z;
        sQt[tid][3] = tn.x; sQt[tid][4] = tn.y; sQt[tid][5] = tn.z;
        sQt[tid][6] = tc.x; sQt[tid][7] = tc.y; sQt[tid][8] = tc.z;
    }
    __syncthreads();

    {   // RBF + frame-projected directions: 8 threads per edge, 36 atom pairs
        int eL  = tid >> 3, sub = tid & 7;
        int e   = e0 + eL;
        int src = eidx[e];
        int tgt = eidx[N_EDGES + e];
        F3 qb = {sQt[eL][0], sQt[eL][1], sQt[eL][2]};
        F3 qn = {sQt[eL][3], sQt[eL][4], sQt[eL][5]};
        F3 qc = {sQt[eL][6], sQt[eL][7], sQt[eL][8]};
        for (int p = sub; p < 36; p += 8) {
            int t = p / 6, s = p % 6;
            F3 xs = ld3(coords + ((long)src * 6 + s) * 3);
            F3 xt = ld3(coords + ((long)tgt * 6 + t) * 3);
            F3 d  = sub3(xs, xt);                  // coords[src][s] - coords[tgt][t]
            float dd = dot3(d, d);
            float D  = sqrtf(dd + FEPS);
            int base = 4 + p * 16;
            #pragma unroll
            for (int k = 0; k < 16; k++) {
                float zq = (D - (20.f / 15.f) * (float)k) * 0.8f;   // /sigma, sigma=1.25
                sfeat[eL][base + k] = f2bf(__expf(-zq * zq));
            }
            F3 u = norm3(d);
            sfeat[eL][580 + p*3 + 0] = f2bf(dot3(u, qb));
            sfeat[eL][580 + p*3 + 1] = f2bf(dot3(u, qn));
            sfeat[eL][580 + p*3 + 2] = f2bf(dot3(u, qc));
        }
        sfeat[eL][688 + sub*2]     = 0;            // K padding 688..703
        sfeat[eL][688 + sub*2 + 1] = 0;
    }
    __syncthreads();

    {   // WMMA GEMM: 16x704 (bf16) @ 704x128 (bf16) -> f32
        const int lane = tid & 31, wave = tid >> 5;
        const int m = lane & 15, g = lane >> 4;
        for (int t = 0; t < 2; t++) {
            const int n0 = wave * 32 + t * 16;
            const int ntile = n0 >> 4;
            v8f c = {};
            for (int ks = 0; ks < 22; ks++) {
                Frag a, b;
                const u16* ap = &sfeat[m][ks * 32];
                #pragma unroll
                for (int i = 0; i < 8; i++) { a.u[i] = ap[g*16 + i]; a.u[8+i] = ap[g*16 + 8 + i]; }
                const u16* bp = &Wsw[((ntile * 22 + ks) * 32 + lane) * 16];
                #pragma unroll
                for (int i = 0; i < 16; i++) b.u[i] = bp[i];
                c = __builtin_amdgcn_wmma_f32_16x16x32_bf16(false, a.v, false, b.v,
                                                            (short)0, c, false, false);
            }
            #pragma unroll
            for (int v = 0; v < 8; v++) {
                int row = g ? (v + 8) : v;
                int col = n0 + m;
                sout[row][col] = c[v] + bias[col];
            }
        }
    }
    __syncthreads();

    if (tid < 16) {
        float s = 0.f, ss = 0.f;
        for (int j = 0; j < HIDDEN; j++) { float y = sout[tid][j]; s += y; ss += y * y; }
        float mu  = s * (1.f / HIDDEN);
        float var = ss * (1.f / HIDDEN) - mu * mu;
        smu[tid] = mu; srstd[tid] = 1.f / sqrtf(fmaxf(var, 0.f) + 1e-5f);
    }
    __syncthreads();
    {
        int row = tid >> 3, part = tid & 7;
        float mu = smu[row], rs = srstd[row];
        long base = (long)(e0 + row) * HIDDEN;
        #pragma unroll
        for (int jj = 0; jj < 16; jj++) {
            int j = part * 16 + jj;
            out[base + j] = (sout[row][j] - mu) * rs * lnw[j] + lnb[j];
        }
    }
}

extern "C" void kernel_launch(void* const* d_in, const int* in_sizes, int n_in,
                              void* d_out, int out_size, void* d_ws, size_t ws_size,
                              hipStream_t stream) {
    (void)in_sizes; (void)n_in; (void)out_size; (void)ws_size;
    const float* coords = (const float*)d_in[0];
    const float* cc     = (const float*)d_in[1];
    const float* nodeW  = (const float*)d_in[2];
    const float* nodeB  = (const float*)d_in[3];
    const float* nodeLw = (const float*)d_in[4];
    const float* nodeLb = (const float*)d_in[5];
    const float* edgeW  = (const float*)d_in[6];
    const float* edgeB  = (const float*)d_in[7];
    const float* edgeLw = (const float*)d_in[8];
    const float* edgeLb = (const float*)d_in[9];
    const int*   batch  = (const int*)d_in[10];
    const int*   eidx   = (const int*)d_in[11];
    float* out = (float*)d_out;

    u16* nodeWsw = (u16*)d_ws;            // 8192 u16
    u16* edgeWsw = nodeWsw + 8192;        // 90112 u16  (total 196608 B of ws)

    convert_weights<<<(8192 + 90112 + 255) / 256, 256, 0, stream>>>(nodeW, edgeW, nodeWsw, edgeWsw);
    node_kernel<<<N_RES / 16, 128, 0, stream>>>(coords, cc, batch, nodeWsw,
                                                nodeB, nodeLw, nodeLb, out);
    edge_kernel<<<N_EDGES / 16, 128, 0, stream>>>(coords, cc, batch, eidx, edgeWsw,
                                                  edgeB, edgeLw, edgeLb,
                                                  out + (long)N_RES * HIDDEN);
}
